// ContrastiveLoss_69226282877397
// MI455X (gfx1250) — matrix operand order repaired and checked by reading, not compiled, not run on hardware
//
#include <hip/hip_runtime.h>
#include <hip/hip_bf16.h>
#include <math.h>

typedef __attribute__((ext_vector_type(16))) __bf16 v16bf;
typedef __attribute__((ext_vector_type(8)))  __bf16 v8bf;
typedef __attribute__((ext_vector_type(8)))  float  v8f;

#define MARGIN 0.2f
#define INV_T  10.0f      // 1/TEMPERATURE
#define KC     128        // K-chunk staged in LDS
#define LDS_STRIDE (KC + 8)   // pad to de-phase LDS banks

// ---------------------------------------------------------------- reductions
__device__ __forceinline__ float blk_sum(float v, float* red, int t, int n) {
  red[t] = v; __syncthreads();
  for (int o = n >> 1; o > 0; o >>= 1) { if (t < o) red[t] += red[t + o]; __syncthreads(); }
  float r = red[0]; __syncthreads();
  return r;
}
__device__ __forceinline__ float blk_max(float v, float* red, int t, int n) {
  red[t] = v; __syncthreads();
  for (int o = n >> 1; o > 0; o >>= 1) { if (t < o) red[t] = fmaxf(red[t], red[t + o]); __syncthreads(); }
  float r = red[0]; __syncthreads();
  return r;
}

// ---------------------------------------------------------------- CDNA5 async copy
__device__ __forceinline__ unsigned lds_off(const void* p) {
  return (unsigned)(size_t)(__attribute__((address_space(3))) const void*)p;
}
// GLOBAL_LOAD_ASYNC_TO_LDS_B128: per-lane 16B memory -> LDS, tracked on ASYNCcnt
__device__ __forceinline__ void async_g2l_b128(unsigned loff, const __bf16* g) {
  asm volatile("global_load_async_to_lds_b128 %0, %1, off"
               :: "v"(loff), "v"((unsigned long long)(size_t)g) : "memory");
}
__device__ __forceinline__ void wait_async0() {
  asm volatile("s_wait_asynccnt 0x0" ::: "memory");
}

// ---------------------------------------------------------------- zero scratch
__global__ void zero_kernel(unsigned* p, int n) {
  for (int i = blockIdx.x * blockDim.x + threadIdx.x; i < n; i += gridDim.x * blockDim.x)
    p[i] = 0u;
}

// ---------------------------------------------------------------- prep
// One block per row b (128 threads):
//  * softmax over sim[b,:] (L=32), caps_mean[b,:] (E=512), normalize -> bf16
//  * h1/h2 rows -> bf16 ; d1[b] = dot(h1[b],h2[b])
//  * qrow[b] = sum (h1-0.5)^2 ; drow[b] = KL row term
__global__ __launch_bounds__(128) void prep_kernel(
    const float* __restrict__ h1, const float* __restrict__ h2,
    const float* __restrict__ caps, const float* __restrict__ sim,
    const float* __restrict__ sv, const float* __restrict__ sv2,
    __bf16* __restrict__ a_b, __bf16* __restrict__ h1b, __bf16* __restrict__ h2b,
    float* __restrict__ d1, float* __restrict__ qrow, float* __restrict__ drow,
    int L, int E, int D) {
  const int b = blockIdx.x;
  const int t = threadIdx.x;
  __shared__ float w[32];
  __shared__ float cm[512];
  __shared__ float red[128];

  if (t < 32) {
    float x = sim[b * L + t];
    float m = x;
    for (int o = 16; o > 0; o >>= 1) m = fmaxf(m, __shfl_xor(m, o, 32));
    float e = __expf(x - m);
    float s = e;
    for (int o = 16; o > 0; o >>= 1) s += __shfl_xor(s, o, 32);
    w[t] = e / s;
  }
  __syncthreads();

  float ssq = 0.f;
  const float* cp = caps + (size_t)b * L * E;
  for (int e = t; e < E; e += 128) {
    float acc = 0.f;
    for (int l = 0; l < L; ++l) acc += w[l] * cp[(size_t)l * E + e];
    cm[e] = acc;
    ssq += acc * acc;
  }
  float nrm2 = blk_sum(ssq, red, t, 128);
  float inv = 1.f / fmaxf(sqrtf(nrm2), 1e-5f);
  for (int e = t; e < E; e += 128) a_b[(size_t)b * E + e] = (__bf16)(cm[e] * inv);
  __syncthreads();

  float h1v = 0.f, h2v = 0.f, xt = -1e30f, xs = -1e30f;
  if (t < D) {
    h1v = h1[(size_t)b * D + t];
    h2v = h2[(size_t)b * D + t];
    h1b[(size_t)b * D + t] = (__bf16)h1v;
    h2b[(size_t)b * D + t] = (__bf16)h2v;
    xt = sv [(size_t)b * D + t] * INV_T;   // teacher
    xs = sv2[(size_t)b * D + t] * INV_T;   // student
  }
  float dv = blk_sum((t < D) ? h1v * h2v : 0.f, red, t, 128);
  float qv = blk_sum((t < D) ? (h1v - 0.5f) * (h1v - 0.5f) : 0.f, red, t, 128);

  float mt = blk_max(xt, red, t, 128);
  float st = blk_sum((t < D) ? __expf(xt - mt) : 0.f, red, t, 128);
  float ms = blk_max(xs, red, t, 128);
  float ss = blk_sum((t < D) ? __expf(xs - ms) : 0.f, red, t, 128);
  float pt = xt - mt - logf(st);
  float ps = xs - ms - logf(ss);
  float contrib = (t < D) ? __expf(pt) * (pt - ps) : 0.f;
  float dl = blk_sum(contrib, red, t, 128);

  if (t == 0) { d1[b] = dv; qrow[b] = qv; drow[b] = dl; }
}

// ---------------------------------------------------------------- WMMA fragments
// A fragment (16x32 bf16, ISA layout: lanes 0-15 hold K in [0,8)+[16,24), lanes 16-31 rest)
__device__ __forceinline__ v16bf frag_a(const __bf16* row, int kbase, int laneHi) {
  v8bf lo = *(const v8bf*)(row + kbase + laneHi * 8);
  v8bf hi = *(const v8bf*)(row + kbase + 16 + laneHi * 8);
  return __builtin_shufflevector(lo, hi, 0,1,2,3,4,5,6,7,8,9,10,11,12,13,14,15);
}
// B fragment (32x16 bf16: lane n holds column n, lanes 0-15 hold K in [0,8)+[16,24), lanes 16-31 rest)
__device__ __forceinline__ v16bf frag_b(const __bf16* row, int kbase, int laneHi) {
  v8bf lo = *(const v8bf*)(row + kbase + laneHi * 8);
  v8bf hi = *(const v8bf*)(row + kbase + 16 + laneHi * 8);
  return __builtin_shufflevector(lo, hi, 0,1,2,3,4,5,6,7,8,9,10,11,12,13,14,15);
}
__device__ __forceinline__ v8f wmma_bf16(v16bf a, v16bf b, v8f c) {
  return __builtin_amdgcn_wmma_f32_16x16x32_bf16(false, a, false, b, (short)0, c, false, false);
}

// cooperative async stage of one K-chunk: A = 32 rows, B = 64 rows, KC elems each.
// Constant trip counts -> straight-line async issues, no exec masking.
__device__ __forceinline__ void stage_chunk(const __bf16* Aglob, const __bf16* Bglob,
                                            __bf16* Albuf, __bf16* Blbuf,
                                            int kc, int E, int tid) {
  const int SEGS = KC / 8;                      // 16B (8 bf16) units per row
#pragma unroll
  for (int r = 0; r < (32 * SEGS) / 256; ++r) { // 2 iterations
    int u = tid + r * 256;
    int row = u / SEGS, seg = u % SEGS;
    async_g2l_b128(lds_off(Albuf + row * LDS_STRIDE + seg * 8),
                   Aglob + (size_t)row * E + kc + seg * 8);
  }
#pragma unroll
  for (int r = 0; r < (64 * SEGS) / 256; ++r) { // 4 iterations
    int u = tid + r * 256;
    int row = u / SEGS, seg = u % SEGS;
    async_g2l_b128(lds_off(Blbuf + row * LDS_STRIDE + seg * 8),
                   Bglob + (size_t)row * E + kc + seg * 8);
  }
}

// ---------------------------------------------------------------- cost passes
// Block (8 waves, 2x4) owns a 32x64 output tile; operands staged in LDS via
// async copies (double-buffered over K). One wave = one 16x16 WMMA tile.
// mode 0: histogram of float bits[31:16]; mode 1: bits[15:0] within selected
// high bin; mode 2: sum of elements with bits < thrBits -> per-block partial.
__global__ __launch_bounds__(256) void cost_pass_kernel(
    const __bf16* __restrict__ a_b, const __bf16* __restrict__ h1b,
    const __bf16* __restrict__ h2b, const float* __restrict__ d1,
    unsigned* __restrict__ hist1, unsigned* __restrict__ hist2,
    const unsigned* __restrict__ scalars, float* __restrict__ partials,
    int blocksJ, int E, int D, int mode) {
  __shared__ __bf16 Abuf[2][32 * LDS_STRIDE];
  __shared__ __bf16 Bbuf[2][64 * LDS_STRIDE];
  __shared__ float red[256];

  const int tid  = threadIdx.x;
  const int wave = tid >> 5;
  const int lane = tid & 31;
  const int laneLow = lane & 15;
  const int laneHi  = lane >> 4;
  const int bi = blockIdx.x / blocksJ;        // 32-row block
  const int bj = blockIdx.x % blocksJ;        // 64-col block
  const int wr = wave >> 2;                   // 0..1
  const int wc = wave & 3;                    // 0..3

  const __bf16* Aglob = a_b + (size_t)(bi * 32) * E;
  const __bf16* Bglob = a_b + (size_t)(bj * 64) * E;

  // prologue: fill buffer 0
  stage_chunk(Aglob, Bglob, Abuf[0], Bbuf[0], 0, E, tid);
  wait_async0();
  __syncthreads();

  v8f acc = {};
  const int nChunks = E / KC;                 // 4
  for (int c = 0; c < nChunks; ++c) {
    if (c + 1 < nChunks)
      stage_chunk(Aglob, Bglob, Abuf[(c + 1) & 1], Bbuf[(c + 1) & 1],
                  (c + 1) * KC, E, tid);      // overlap copy with compute
    const __bf16* Ab = Abuf[c & 1] + (size_t)(wr * 16 + laneLow) * LDS_STRIDE;
    const __bf16* Bb = Bbuf[c & 1] + (size_t)(wc * 16 + laneLow) * LDS_STRIDE;
#pragma unroll
    for (int ks = 0; ks < KC; ks += 32)
      acc = wmma_bf16(frag_a(Ab, ks, laneHi), frag_b(Bb, ks, laneHi), acc);
    wait_async0();
    __syncthreads();
  }

  // scores = h1 @ h2.T  (K = 64, register-direct)
  const __bf16* h1r = h1b + (size_t)(bi * 32 + wr * 16 + laneLow) * D;
  const __bf16* h2r = h2b + (size_t)(bj * 64 + wc * 16 + laneLow) * D;
  v8f accs = {};
#pragma unroll
  for (int kb = 0; kb < 64; kb += 32)
    accs = wmma_bf16(frag_a(h1r, kb, laneHi), frag_b(h2r, kb, laneHi), accs);

  const unsigned selBin  = scalars[0];
  const unsigned thrBits = scalars[2];
  const int i0 = bi * 32 + wr * 16;
  const int j0 = bj * 64 + wc * 16;
  float lsum = 0.f;
#pragma unroll
  for (int v = 0; v < 8; ++v) {
    int i = i0 + v + 8 * laneHi;              // C/D layout: lanes16-31 -> M = v+8
    int j = j0 + laneLow;
    float cosv = fminf(fmaxf(acc[v], 1e-6f), 1.f - 1e-6f);
    float cost = (1.f - cosv) * MARGIN + accs[v] - d1[i];
    cost = fmaxf(cost, 1e-8f);
    if (i == j) cost = 0.f;
    unsigned bits = __float_as_uint(cost);    // cost >= 0 -> uint ordering valid
    if (mode == 0) {
      atomicAdd(&hist1[bits >> 16], 1u);
    } else if (mode == 1) {
      if ((bits >> 16) == selBin) atomicAdd(&hist2[bits & 0xFFFFu], 1u);
    } else {
      if (bits < thrBits) lsum += cost;
    }
  }
  if (mode == 2) {
    float s = blk_sum(lsum, red, tid, 256);
    if (tid == 0) partials[blockIdx.x] = s;
  }
}

// ---------------------------------------------------------------- radix-select scan
__global__ __launch_bounds__(256) void scan_kernel(const unsigned* __restrict__ hist,
                                                   unsigned* __restrict__ scalars,
                                                   const int* __restrict__ epochs,
                                                   double totalElems, int phase) {
  __shared__ unsigned sums[256];
  __shared__ unsigned pre[257];
  const int t = threadIdx.x;
  const int base = t * 256;
  unsigned local = 0;
  for (int i = 0; i < 256; ++i) local += hist[base + i];
  sums[t] = local;
  __syncthreads();
  if (t == 0) {
    unsigned run = 0;
    for (int i = 0; i < 256; ++i) { pre[i] = run; run += sums[i]; }
    pre[256] = run;
  }
  __syncthreads();
  unsigned k;
  if (phase == 0) {
    double cr = 0.1 * (double)(*epochs);
    double clean = (cr < 1.0) ? cr : 0.99;
    k = (unsigned)ceil(totalElems * (1.0 - clean));
  } else {
    k = scalars[1];
  }
  if (pre[t] < k && pre[t] + sums[t] >= k) {   // exactly one thread
    unsigned cum = pre[t];
    for (int i = 0; i < 256; ++i) {
      unsigned c = hist[base + i];
      if (cum + c >= k) {
        if (phase == 0) { scalars[0] = (unsigned)(base + i); scalars[1] = k - cum; }
        else            { scalars[2] = (scalars[0] << 16) | (unsigned)(base + i); }
        break;
      }
      cum += c;
    }
  }
}

// ---------------------------------------------------------------- final combine
__global__ __launch_bounds__(256) void final_kernel(const float* __restrict__ partials,
                                                    const float* __restrict__ qrow,
                                                    const float* __restrict__ drow,
                                                    float* __restrict__ out,
                                                    int nPart, int B, int nbits) {
  __shared__ float red[256];
  const int t = threadIdx.x;
  float s1 = 0.f, s2 = 0.f, s3 = 0.f;
  for (int i = t; i < nPart; i += 256) s1 += partials[i];
  for (int i = t; i < B; i += 256) { s2 += qrow[i]; s3 += drow[i]; }
  float L1 = blk_sum(s1, red, t, 256);
  float Q  = blk_sum(s2, red, t, 256);
  float L3 = blk_sum(s3, red, t, 256);
  if (t == 0) {
    float L2 = -Q / ((float)nbits * (float)B);
    out[0] = 0.5f * L1 + 0.5f * L2 + 0.5f * L3 + 3.f * logf(2.f);
  }
}

// ---------------------------------------------------------------- launch
extern "C" void kernel_launch(void* const* d_in, const int* in_sizes, int n_in,
                              void* d_out, int out_size, void* d_ws, size_t ws_size,
                              hipStream_t stream) {
  const float* h1   = (const float*)d_in[0];
  const float* h2   = (const float*)d_in[1];
  const float* caps = (const float*)d_in[2];
  const float* sim  = (const float*)d_in[3];
  const float* sv   = (const float*)d_in[4];
  const float* sv2  = (const float*)d_in[5];
  const int* epochs = (const int*)d_in[6];

  const int D = 64, L = 32, E = 512, NBITS = 64;
  const int B = in_sizes[0] / D;                 // 4096
  const int blocksI = B / 32;                    // 128
  const int blocksJ = B / 64;                    // 64
  const int gemmBlocks = blocksI * blocksJ;      // 8192

  // workspace layout (256B-aligned chunks)
  char* ws = (char*)d_ws;
  size_t off = 0;
  auto carve = [&](size_t bytes) { size_t o = off; off += (bytes + 255) & ~(size_t)255; return o; };
  __bf16* a_b  = (__bf16*)(ws + carve((size_t)B * E * 2));
  __bf16* h1b  = (__bf16*)(ws + carve((size_t)B * D * 2));
  __bf16* h2b  = (__bf16*)(ws + carve((size_t)B * D * 2));
  float*  d1   = (float*) (ws + carve((size_t)B * 4));
  float*  qrow = (float*) (ws + carve((size_t)B * 4));
  float*  drow = (float*) (ws + carve((size_t)B * 4));
  size_t zstart = off;                           // zero from here down
  unsigned* hist1   = (unsigned*)(ws + carve(65536 * 4));
  unsigned* hist2   = (unsigned*)(ws + carve(65536 * 4));
  unsigned* scalars = (unsigned*)(ws + carve(64));
  float*    parts   = (float*)   (ws + carve((size_t)gemmBlocks * 4));
  size_t zwords = (off - zstart) / 4;

  // 1) zero histograms / scalars / partials
  zero_kernel<<<512, 256, 0, stream>>>((unsigned*)(ws + zstart), (int)zwords);

  // 2) per-row prep
  prep_kernel<<<B, 128, 0, stream>>>(h1, h2, caps, sim, sv, sv2,
                                     a_b, h1b, h2b, d1, qrow, drow, L, E, D);

  // 3) WMMA pass 0: high-16-bit histogram of cost matrix
  cost_pass_kernel<<<gemmBlocks, 256, 0, stream>>>(a_b, h1b, h2b, d1, hist1, hist2,
                                                   scalars, parts, blocksJ, E, D, 0);
  // 4) select high bin + residual rank
  scan_kernel<<<1, 256, 0, stream>>>(hist1, scalars, epochs, (double)B * (double)B, 0);
  // 5) WMMA pass 1: low-16-bit histogram within selected bin
  cost_pass_kernel<<<gemmBlocks, 256, 0, stream>>>(a_b, h1b, h2b, d1, hist1, hist2,
                                                   scalars, parts, blocksJ, E, D, 1);
  // 6) exact k-th-smallest bit pattern
  scan_kernel<<<1, 256, 0, stream>>>(hist2, scalars, epochs, (double)B * (double)B, 1);
  // 7) WMMA pass 2: sum of cost elements strictly below threshold
  cost_pass_kernel<<<gemmBlocks, 256, 0, stream>>>(a_b, h1b, h2b, d1, hist1, hist2,
                                                   scalars, parts, blocksJ, E, D, 2);
  // 8) deterministic final combine
  final_kernel<<<1, 256, 0, stream>>>(parts, qrow, drow, (float*)d_out,
                                      gemmBlocks, B, NBITS);
}
